// SSM_91164975825358
// MI455X (gfx1250) — matrix operand order repaired and checked by reading, not compiled, not run on hardware
//
#include <hip/hip_runtime.h>
#include <hip/hip_bf16.h>
#include <math.h>

// ---------------- problem constants ----------------
#define B_C      2
#define S_C      2048
#define E_C      2048
#define H_C      64
#define P_C      64
#define N_C      128
#define K_C      4
#define CH_C     256
#define NC_C     8          // S / CH
#define INTER_C  4096       // H*P
#define CONVD_C  4352       // INTER + 2*N
#define PROJ_C   8512       // INTER + CONVD + H
#define PROJ_LD  8576       // padded to multiple of 128
#define EPS_C    1e-5f

// ---------------- CDNA5 async global->LDS (guarded; falls back cleanly) ----
#if defined(__AMDGCN__) && __has_builtin(__builtin_amdgcn_global_load_async_to_lds_b128) && __has_builtin(__builtin_amdgcn_s_wait_asynccnt)
#define USE_ASYNC_LDS 1
#else
#define USE_ASYNC_LDS 0
#endif

// ---------------- WMMA types / helpers ----------------
typedef __bf16 bf16_t;
typedef __attribute__((ext_vector_type(16))) __bf16 bf16x16;
typedef __attribute__((ext_vector_type(8)))  __bf16 bf16x8;
typedef __attribute__((ext_vector_type(8)))  float  f32x8;
typedef int v4i __attribute__((ext_vector_type(4)));
typedef __attribute__((address_space(1))) v4i g_v4i;   // global (device) int4
typedef __attribute__((address_space(3))) v4i l_v4i;   // LDS int4

struct __align__(16) U4 { unsigned int x, y, z, w; };

union FragU { bf16x16 v; bf16x8 h[2]; };

// A-matrix fragment (16 rows x 32 K, bf16). Per ISA 7.12.2:
// lanes 0-15: row=lane, K = {0..7} and {16..23}; lanes 16-31: row=lane-16, K={8..15},{24..31}
__device__ __forceinline__ bf16x16 load_fragA(const bf16_t* base, int pitch) {
  const int lane = threadIdx.x & 31;
  const bf16_t* p = base + (lane & 15) * pitch + ((lane >> 4) << 3);
  FragU f;
  f.h[0] = *(const bf16x8*)(p);
  f.h[1] = *(const bf16x8*)(p + 16);
  return f.v;
}
// B-matrix fragment (32 K x 16 cols) from B^T storage (row n holds K-contiguous data):
// lanes 0-15: col=lane, K = {0..7} and {16..23}; lanes 16-31: col=lane-16, K={8..15},{24..31}
__device__ __forceinline__ bf16x16 load_fragB(const bf16_t* base, int pitch) {
  const int lane = threadIdx.x & 31;
  const bf16_t* p = base + (lane & 15) * pitch + ((lane >> 4) << 3);
  FragU f;
  f.h[0] = *(const bf16x8*)(p);
  f.h[1] = *(const bf16x8*)(p + 16);
  return f.v;
}
__device__ __forceinline__ f32x8 wmma_bf16(bf16x16 a, bf16x16 b, f32x8 c) {
  return __builtin_amdgcn_wmma_f32_16x16x32_bf16(false, a, false, b, (short)0, c, false, false);
}
__device__ __forceinline__ bf16_t f2bf(float x) { return (bf16_t)x; }
__device__ __forceinline__ float sigmoidf_(float x) { return 1.f / (1.f + __expf(-x)); }

// ---------------- elementwise / conversion kernels ----------------
__global__ __launch_bounds__(256) void ssm_cvt_x_mask(
    const float* __restrict__ x, const int* __restrict__ mask, bf16_t* __restrict__ xb) {
  size_t i = (size_t)blockIdx.x * 256 + threadIdx.x;
  size_t total = (size_t)B_C * S_C * E_C;
  if (i >= total) return;
  size_t bs = i / E_C;
  int s = (int)(bs % S_C);
  int b = (int)(bs / S_C);
  float pm = (mask[(size_t)(b * 2 + 1) * S_C + s] == 0) ? 1.f : 0.f;
  xb[i] = f2bf(x[i] * pm);
}

__global__ __launch_bounds__(256) void ssm_cvt_bf16(
    const float* __restrict__ src, bf16_t* __restrict__ dst, size_t n) {
  size_t i = (size_t)blockIdx.x * 256 + threadIdx.x;
  if (i < n) dst[i] = f2bf(src[i]);
}

// ---------------- generic batched WMMA GEMM: C = A * B^T ----------------
// A: [M,K] bf16 row-major (pitch K), B: [N,K] bf16 row-major (pitch K), C: f32 pitch ldc
// Block tile 256x128x32, 8 waves as 4(M) x 2(N), each wave 64x64 = 16 WMMA per K-step.
// Double-buffered LDS; async global->LDS DMA when available (ASYNCcnt-tracked).
#define GBM 256
#define GBN 128
#define GBK 32
#define GLDP 40           // GBK + 8 elements; 80B row pitch (multiple of 16B)
#define A_U4 4            // GBM*GBK/8 / 256 threads
#define B_U4 2            // GBN*GBK/8 / 256 threads

__global__ __launch_bounds__(256) void ssm_gemm_bf16_tn(
    const bf16_t* __restrict__ A, const bf16_t* __restrict__ Bw, float* __restrict__ C,
    int M, int N, int K, int ldc,
    long long strideA, long long strideB, long long strideC) {
  __shared__ bf16_t As[2][GBM * GLDP];
  __shared__ bf16_t Bs[2][GBN * GLDP];
  const int tid = threadIdx.x;
  const int wid = tid >> 5;
  const int bm = blockIdx.x * GBM;
  const int bn = blockIdx.y * GBN;
  A  += (size_t)blockIdx.z * strideA;
  Bw += (size_t)blockIdx.z * strideB;
  C  += (size_t)blockIdx.z * strideC;
  const int wm = (wid >> 1) * 64;   // 0,64,128,192
  const int wn = (wid & 1) * 64;    // 0,64

  f32x8 acc[4][4];
  f32x8 zero = {0.f,0.f,0.f,0.f,0.f,0.f,0.f,0.f};
  #pragma unroll
  for (int i = 0; i < 4; ++i)
    #pragma unroll
    for (int j = 0; j < 4; ++j) acc[i][j] = zero;

  const int nk = K / GBK;

#if USE_ASYNC_LDS
  // -------- async DMA pipeline --------
  #pragma unroll
  for (int i = 0; i < A_U4; ++i) {           // prologue: tile 0 -> buffer 0
    int u = tid + 256 * i;
    int row = u >> 2, cu = (u & 3) * 8;
    int gr = bm + row; if (gr > M - 1) gr = M - 1;
    __builtin_amdgcn_global_load_async_to_lds_b128(
        (g_v4i*)(A + (size_t)gr * K + cu),
        (l_v4i*)(&As[0][row * GLDP + cu]), 0, 0);
  }
  #pragma unroll
  for (int i = 0; i < B_U4; ++i) {
    int u = tid + 256 * i;
    int row = u >> 2, cu = (u & 3) * 8;
    int gr = bn + row; if (gr > N - 1) gr = N - 1;
    __builtin_amdgcn_global_load_async_to_lds_b128(
        (g_v4i*)(Bw + (size_t)gr * K + cu),
        (l_v4i*)(&Bs[0][row * GLDP + cu]), 0, 0);
  }
  for (int k = 0; k < nk; ++k) {
    const int cb = k & 1;
    if (k + 1 < nk) {
      const int nb = cb ^ 1;
      const int kbase = (k + 1) * GBK;
      #pragma unroll
      for (int i = 0; i < A_U4; ++i) {
        int u = tid + 256 * i;
        int row = u >> 2, cu = (u & 3) * 8;
        int gr = bm + row; if (gr > M - 1) gr = M - 1;
        __builtin_amdgcn_global_load_async_to_lds_b128(
            (g_v4i*)(A + (size_t)gr * K + kbase + cu),
            (l_v4i*)(&As[nb][row * GLDP + cu]), 0, 0);
      }
      #pragma unroll
      for (int i = 0; i < B_U4; ++i) {
        int u = tid + 256 * i;
        int row = u >> 2, cu = (u & 3) * 8;
        int gr = bn + row; if (gr > N - 1) gr = N - 1;
        __builtin_amdgcn_global_load_async_to_lds_b128(
            (g_v4i*)(Bw + (size_t)gr * K + kbase + cu),
            (l_v4i*)(&Bs[nb][row * GLDP + cu]), 0, 0);
      }
      // 6 ops just issued may stay outstanding; in-order completion => tile k landed
      __builtin_amdgcn_s_wait_asynccnt(6);
    } else {
      __builtin_amdgcn_s_wait_asynccnt(0);
    }
    __syncthreads();
    bf16x16 af[4], bfx[4];
    #pragma unroll
    for (int i = 0; i < 4; ++i) af[i] = load_fragA(&As[cb][(wm + i * 16) * GLDP], GLDP);
    #pragma unroll
    for (int j = 0; j < 4; ++j) bfx[j] = load_fragB(&Bs[cb][(wn + j * 16) * GLDP], GLDP);
    #pragma unroll
    for (int i = 0; i < 4; ++i)
      #pragma unroll
      for (int j = 0; j < 4; ++j)
        acc[i][j] = wmma_bf16(af[i], bfx[j], acc[i][j]);
    __syncthreads();   // protect buffer reuse before next issue round
  }
#else
  // -------- register-staged double-buffered fallback --------
  U4 ra[A_U4], rb4[B_U4];
  #pragma unroll
  for (int i = 0; i < A_U4; ++i) {
    int u = tid + 256 * i;
    int row = u >> 2, cu = (u & 3) * 8;
    int gr = bm + row; if (gr > M - 1) gr = M - 1;
    ra[i] = *(const U4*)(A + (size_t)gr * K + cu);
  }
  #pragma unroll
  for (int i = 0; i < B_U4; ++i) {
    int u = tid + 256 * i;
    int row = u >> 2, cu = (u & 3) * 8;
    int gr = bn + row; if (gr > N - 1) gr = N - 1;
    rb4[i] = *(const U4*)(Bw + (size_t)gr * K + cu);
  }
  #pragma unroll
  for (int i = 0; i < A_U4; ++i) {
    int u = tid + 256 * i;
    int row = u >> 2, cu = (u & 3) * 8;
    *(U4*)(&As[0][row * GLDP + cu]) = ra[i];
  }
  #pragma unroll
  for (int i = 0; i < B_U4; ++i) {
    int u = tid + 256 * i;
    int row = u >> 2, cu = (u & 3) * 8;
    *(U4*)(&Bs[0][row * GLDP + cu]) = rb4[i];
  }
  for (int k = 0; k < nk; ++k) {
    const int cb = k & 1;
    const bool more = (k + 1 < nk);
    if (more) {
      const int kbase = (k + 1) * GBK;
      #pragma unroll
      for (int i = 0; i < A_U4; ++i) {
        int u = tid + 256 * i;
        int row = u >> 2, cu = (u & 3) * 8;
        int gr = bm + row; if (gr > M - 1) gr = M - 1;
        ra[i] = *(const U4*)(A + (size_t)gr * K + kbase + cu);
      }
      #pragma unroll
      for (int i = 0; i < B_U4; ++i) {
        int u = tid + 256 * i;
        int row = u >> 2, cu = (u & 3) * 8;
        int gr = bn + row; if (gr > N - 1) gr = N - 1;
        rb4[i] = *(const U4*)(Bw + (size_t)gr * K + kbase + cu);
      }
    }
    __syncthreads();
    bf16x16 af[4], bfx[4];
    #pragma unroll
    for (int i = 0; i < 4; ++i) af[i] = load_fragA(&As[cb][(wm + i * 16) * GLDP], GLDP);
    #pragma unroll
    for (int j = 0; j < 4; ++j) bfx[j] = load_fragB(&Bs[cb][(wn + j * 16) * GLDP], GLDP);
    #pragma unroll
    for (int i = 0; i < 4; ++i)
      #pragma unroll
      for (int j = 0; j < 4; ++j)
        acc[i][j] = wmma_bf16(af[i], bfx[j], acc[i][j]);
    __syncthreads();
    if (more) {
      const int nb = cb ^ 1;
      #pragma unroll
      for (int i = 0; i < A_U4; ++i) {
        int u = tid + 256 * i;
        int row = u >> 2, cu = (u & 3) * 8;
        *(U4*)(&As[nb][row * GLDP + cu]) = ra[i];
      }
      #pragma unroll
      for (int i = 0; i < B_U4; ++i) {
        int u = tid + 256 * i;
        int row = u >> 2, cu = (u & 3) * 8;
        *(U4*)(&Bs[nb][row * GLDP + cu]) = rb4[i];
      }
    }
  }
#endif

  // store C (D-matrix layout: VGPR r -> row r (lanes 0-15) / row r+8 (lanes 16-31))
  const int lane = tid & 31;
  const int col = lane & 15;
  const int rb = (lane >> 4) << 3;
  for (int i = 0; i < 4; ++i)
    for (int j = 0; j < 4; ++j) {
      int n0 = bn + wn + j * 16 + col;
      if (n0 >= N) continue;
      int m0 = bm + wm + i * 16 + rb;
      #pragma unroll
      for (int r = 0; r < 8; ++r) {
        int m = m0 + r;
        if (m < M) C[(size_t)m * ldc + n0] = acc[i][j][r];
      }
    }
}

// ---------------- causal conv1d (K=4) + bias + SiLU + mask ----------------
__global__ __launch_bounds__(256) void ssm_conv_silu(
    const float* __restrict__ proj, const float* __restrict__ conv_w,
    const float* __restrict__ conv_b, const int* __restrict__ mask,
    float* __restrict__ hBC) {
  size_t i = (size_t)blockIdx.x * 256 + threadIdx.x;
  size_t total = (size_t)B_C * S_C * CONVD_C;
  if (i >= total) return;
  int c = (int)(i % CONVD_C);
  size_t bs = i / CONVD_C;
  int s = (int)(bs % S_C);
  int b = (int)(bs / S_C);
  float acc = conv_b[c];
  #pragma unroll
  for (int k = 0; k < K_C; ++k) {
    int sp = s - (K_C - 1) + k;
    if (sp >= 0)
      acc += proj[((size_t)b * S_C + sp) * PROJ_LD + INTER_C + c] * conv_w[c * K_C + k];
  }
  acc = acc * sigmoidf_(acc);
  float pm = (mask[(size_t)(b * 2 + 1) * S_C + s] == 0) ? 1.f : 0.f;
  hBC[i] = acc * pm;
}

// ---------------- dt = softplus(dt+bias), dA cumsum per chunk ----------------
__global__ __launch_bounds__(256) void ssm_dt_scan(
    const float* __restrict__ proj, const float* __restrict__ dt_bias,
    const float* __restrict__ A_log,
    float* __restrict__ dt_g, float* __restrict__ Acum, float* __restrict__ Alast) {
  int idx = blockIdx.x;                 // b*H*NC
  int c = idx % NC_C;
  int h = (idx / NC_C) % H_C;
  int b = idx / (NC_C * H_C);
  int l = threadIdx.x;                  // 0..255
  int s = c * CH_C + l;
  float raw = proj[((size_t)b * S_C + s) * PROJ_LD + INTER_C + CONVD_C + h] + dt_bias[h];
  float dt = (raw > 20.f) ? raw : log1pf(__expf(raw));
  if (dt < 0.f) dt = 0.f;
  dt_g[((size_t)b * S_C + s) * H_C + h] = dt;
  float a = -__expf(A_log[h]);
  __shared__ float sm[CH_C];
  sm[l] = a * dt;
  __syncthreads();
  for (int off = 1; off < CH_C; off <<= 1) {
    float v = (l >= off) ? sm[l - off] : 0.f;
    __syncthreads();
    sm[l] += v;
    __syncthreads();
  }
  size_t base = ((size_t)(b * H_C + h) * NC_C + c);
  Acum[base * CH_C + l] = sm[l];
  if (l == CH_C - 1) Alast[base] = sm[l];
}

// ---------------- bf16 copies of B,C chunks + B^T ----------------
__global__ __launch_bounds__(256) void ssm_prep_bc(
    const float* __restrict__ hBC, bf16_t* __restrict__ Bb,
    bf16_t* __restrict__ Cb, bf16_t* __restrict__ Bt) {
  int bc = blockIdx.x;                 // b*NC + c
  int c = bc % NC_C;
  int b = bc / NC_C;
  size_t outb = (size_t)bc * CH_C * N_C;
  for (int i = threadIdx.x; i < CH_C * N_C; i += 256) {
    int l = i >> 7;                    // /N_C
    int n = i & (N_C - 1);
    const float* row = hBC + ((size_t)b * S_C + c * CH_C + l) * CONVD_C + INTER_C;
    bf16_t bb = f2bf(row[n]);
    Bb[outb + i] = bb;
    Cb[outb + i] = f2bf(row[N_C + n]);
    Bt[outb + (size_t)n * CH_C + l] = bb;
  }
}

// ---------------- per-chunk states: states[p,n] = sum_l X[l,p]*dt[l]*decay[l] * B[l,n] ----------------
#define XT_P (CH_C + 8)   // 264 elements, 528B (mult of 16)
__global__ __launch_bounds__(256) void ssm_chunk_states(
    const float* __restrict__ hBC, const float* __restrict__ dt_g,
    const float* __restrict__ Acum, const float* __restrict__ Alast,
    const bf16_t* __restrict__ Bt16, float* __restrict__ states_c) {
  int idx = blockIdx.x;                // b*NC*H
  int h = idx % H_C;
  int c = (idx / H_C) % NC_C;
  int b = idx / (H_C * NC_C);
  __shared__ bf16_t Xt[P_C * XT_P];
  __shared__ float scale_s[CH_C];
  const int tid = threadIdx.x;
  {
    int l = tid;
    size_t abase = ((size_t)(b * H_C + h) * NC_C + c);
    float al = Alast[abase];
    float ac = Acum[abase * CH_C + l];
    float d  = dt_g[((size_t)b * S_C + c * CH_C + l) * H_C + h];
    scale_s[l] = __expf(al - ac) * d;
  }
  __syncthreads();
  const float* Xg = hBC + ((size_t)b * S_C + c * CH_C) * CONVD_C + h * P_C;
  for (int i = tid; i < CH_C * P_C; i += 256) {
    int l = i >> 6;
    int p = i & 63;
    Xt[p * XT_P + l] = f2bf(Xg[(size_t)l * CONVD_C + p] * scale_s[l]);
  }
  __syncthreads();
  const int wid = tid >> 5;
  const int wm = (wid >> 1) * 16;      // 0..48 over P
  const int wn = (wid & 1) * 64;       // 0/64 over N
  const bf16_t* Bt = Bt16 + (size_t)(b * NC_C + c) * N_C * CH_C;
  f32x8 acc[4];
  f32x8 zero = {0.f,0.f,0.f,0.f,0.f,0.f,0.f,0.f};
  for (int j = 0; j < 4; ++j) acc[j] = zero;
  for (int k0 = 0; k0 < CH_C; k0 += 32) {
    bf16x16 a = load_fragA(&Xt[wm * XT_P + k0], XT_P);
    #pragma unroll
    for (int j = 0; j < 4; ++j) {
      bf16x16 bfr = load_fragB(Bt + (size_t)(wn + j * 16) * CH_C + k0, CH_C);
      acc[j] = wmma_bf16(a, bfr, acc[j]);
    }
  }
  float* out = states_c + ((size_t)(b * NC_C + c) * H_C + h) * P_C * N_C;
  const int lane = tid & 31;
  const int col = lane & 15;
  const int rb = (lane >> 4) << 3;
  for (int j = 0; j < 4; ++j) {
    int n0 = wn + j * 16 + col;
    int m0 = wm + rb;
    #pragma unroll
    for (int r = 0; r < 8; ++r)
      out[(size_t)(m0 + r) * N_C + n0] = acc[j][r];
  }
}

// ---------------- inter-chunk recurrence -> bf16 prev states ----------------
__global__ __launch_bounds__(256) void ssm_recurrence(
    const float* __restrict__ states_c, const float* __restrict__ Alast,
    bf16_t* __restrict__ prev16) {
  size_t i = (size_t)blockIdx.x * 256 + threadIdx.x;   // over B*H*P*N
  int n = (int)(i % N_C);
  int p = (int)((i / N_C) % P_C);
  int h = (int)((i / ((size_t)N_C * P_C)) % H_C);
  int b = (int)(i / ((size_t)N_C * P_C * H_C));
  float run = 0.f;
  for (int c = 0; c < NC_C; ++c) {
    size_t off = (((size_t)(b * NC_C + c) * H_C + h) * P_C + p) * N_C + n;
    prev16[off] = f2bf(run);
    run = run * __expf(Alast[(size_t)(b * H_C + h) * NC_C + c]) + states_c[off];
  }
}

// ---------------- Y = (L*CB^T) @ (X*dt) + exp(Acum)*C @ prev^T + D*X ----------------
#define YALD 40   // 32 + 8
__global__ __launch_bounds__(256) void ssm_y_kernel(
    const float* __restrict__ hBC, const float* __restrict__ dt_g,
    const float* __restrict__ Acum, const float* __restrict__ CBt,
    const bf16_t* __restrict__ prev16, const float* __restrict__ Dv,
    float* __restrict__ y_g) {
  int idx = blockIdx.x;                // b*NC*H
  int h = idx % H_C;
  int c = (idx / H_C) % NC_C;
  int b = idx / (H_C * NC_C);
  __shared__ bf16_t Xt[P_C * XT_P];    // X*dt transposed: [p][s]
  __shared__ bf16_t As[CH_C * YALD];   // staged A tiles (256 x 32)
  __shared__ float ac_s[CH_C];
  const int tid = threadIdx.x;
  {
    int l = tid;
    ac_s[l] = Acum[((size_t)(b * H_C + h) * NC_C + c) * CH_C + l];
  }
  __syncthreads();
  const float* Xg = hBC + ((size_t)b * S_C + c * CH_C) * CONVD_C + h * P_C;
  for (int i = tid; i < CH_C * P_C; i += 256) {
    int l = i >> 6;
    int p = i & 63;
    float d = dt_g[((size_t)b * S_C + c * CH_C + l) * H_C + h];
    Xt[p * XT_P + l] = f2bf(Xg[(size_t)l * CONVD_C + p] * d);
  }

  const int wid = tid >> 5;
  const int wm = wid * 32;             // 0..224 over L
  f32x8 acc[2][4];
  f32x8 zero = {0.f,0.f,0.f,0.f,0.f,0.f,0.f,0.f};
  for (int i = 0; i < 2; ++i) for (int j = 0; j < 4; ++j) acc[i][j] = zero;

  const float* CBt_bc = CBt + (size_t)(b * NC_C + c) * CH_C * CH_C;
  // phase 1: K over s (causal-masked decayed CB^T)
  for (int s0 = 0; s0 < CH_C; s0 += 32) {
    __syncthreads();
    for (int i = tid; i < CH_C * 32; i += 256) {
      int l = i >> 5;
      int s = s0 + (i & 31);
      float v = 0.f;
      if (s <= l) v = CBt_bc[(size_t)l * CH_C + s] * __expf(ac_s[l] - ac_s[s]);
      As[l * YALD + (i & 31)] = f2bf(v);
    }
    __syncthreads();
    bf16x16 af[2];
    #pragma unroll
    for (int i = 0; i < 2; ++i) af[i] = load_fragA(&As[(wm + i * 16) * YALD], YALD);
    #pragma unroll
    for (int j = 0; j < 4; ++j) {
      bf16x16 bfr = load_fragB(&Xt[(j * 16) * XT_P + s0], XT_P);
      #pragma unroll
      for (int i = 0; i < 2; ++i) acc[i][j] = wmma_bf16(af[i], bfr, acc[i][j]);
    }
  }
  // phase 2: K over n: A = exp(Acum[l]) * C[l,n], B = prev^T rows p (K=n contiguous)
  const float* Cg = hBC + ((size_t)b * S_C + c * CH_C) * CONVD_C + INTER_C + N_C;
  const bf16_t* Pg = prev16 + ((size_t)(b * NC_C + c) * H_C + h) * P_C * N_C;
  for (int n0 = 0; n0 < N_C; n0 += 32) {
    __syncthreads();
    for (int i = tid; i < CH_C * 32; i += 256) {
      int l = i >> 5;
      int nn = i & 31;
      float v = Cg[(size_t)l * CONVD_C + n0 + nn] * __expf(ac_s[l]);
      As[l * YALD + nn] = f2bf(v);
    }
    __syncthreads();
    bf16x16 af[2];
    #pragma unroll
    for (int i = 0; i < 2; ++i) af[i] = load_fragA(&As[(wm + i * 16) * YALD], YALD);
    #pragma unroll
    for (int j = 0; j < 4; ++j) {
      bf16x16 bfr = load_fragB(Pg + (size_t)(j * 16) * N_C + n0, N_C);
      #pragma unroll
      for (int i = 0; i < 2; ++i) acc[i][j] = wmma_bf16(af[i], bfr, acc[i][j]);
    }
  }
  // epilogue: + D*X, store to y_g (b,s,INTER)
  const int lane = tid & 31;
  const int col = lane & 15;
  const int rb = (lane >> 4) << 3;
  float Dh = Dv[h];
  for (int i = 0; i < 2; ++i)
    for (int j = 0; j < 4; ++j) {
      int pcol = j * 16 + col;
      #pragma unroll
      for (int r = 0; r < 8; ++r) {
        int l = wm + i * 16 + rb + r;
        float xr = Xg[(size_t)l * CONVD_C + pcol];
        y_g[((size_t)b * S_C + c * CH_C + l) * INTER_C + h * P_C + pcol] =
            acc[i][j][r] + Dh * xr;
      }
    }
}

// ---------------- gate * silu + RMSNorm -> bf16 ----------------
__global__ __launch_bounds__(256) void ssm_gate_norm(
    const float* __restrict__ y_g, const float* __restrict__ proj,
    const float* __restrict__ norm_w, bf16_t* __restrict__ yb) {
  int row = blockIdx.x;                // b*S
  const float* y = y_g + (size_t)row * INTER_C;
  const float* g = proj + (size_t)row * PROJ_LD;
  const int tid = threadIdx.x;
  float ss = 0.f;
  for (int i = tid; i < INTER_C; i += 256) {
    float gv = g[i];
    float hs = y[i] * gv * sigmoidf_(gv);
    ss += hs * hs;
  }
  __shared__ float red[256];
  red[tid] = ss;
  __syncthreads();
  for (int off = 128; off > 0; off >>= 1) {
    if (tid < off) red[tid] += red[tid + off];
    __syncthreads();
  }
  float rs = rsqrtf(red[0] / (float)INTER_C + EPS_C);
  for (int i = tid; i < INTER_C; i += 256) {
    float gv = g[i];
    float hs = y[i] * gv * sigmoidf_(gv);
    yb[(size_t)row * INTER_C + i] = f2bf(norm_w[i] * hs * rs);
  }
}

// ---------------- launch ----------------
static inline void* wsalloc(char*& cur, size_t bytes) {
  void* p = (void*)cur;
  cur += (bytes + 255) & ~(size_t)255;
  return p;
}

extern "C" void kernel_launch(void* const* d_in, const int* in_sizes, int n_in,
                              void* d_out, int out_size, void* d_ws, size_t ws_size,
                              hipStream_t stream) {
  (void)in_sizes; (void)n_in; (void)out_size; (void)ws_size;
  const float* x        = (const float*)d_in[0];
  const int*   mask     = (const int*)d_in[1];
  const float* in_w     = (const float*)d_in[2];
  const float* conv_w   = (const float*)d_in[3];
  const float* conv_b   = (const float*)d_in[4];
  const float* dt_bias  = (const float*)d_in[5];
  const float* A_log    = (const float*)d_in[6];
  const float* Dv       = (const float*)d_in[7];
  const float* norm_w   = (const float*)d_in[8];
  const float* out_w    = (const float*)d_in[9];
  float* out = (float*)d_out;

  char* cur = (char*)d_ws;
  bf16_t* xb      = (bf16_t*)wsalloc(cur, (size_t)B_C * S_C * E_C * 2);
  bf16_t* w_in_b  = (bf16_t*)wsalloc(cur, (size_t)PROJ_C * E_C * 2);
  bf16_t* w_out_b = (bf16_t*)wsalloc(cur, (size_t)E_C * INTER_C * 2);
  float*  proj    = (float*) wsalloc(cur, (size_t)B_C * S_C * PROJ_LD * 4);
  float*  hBC     = (float*) wsalloc(cur, (size_t)B_C * S_C * CONVD_C * 4);
  float*  dt_g    = (float*) wsalloc(cur, (size_t)B_C * S_C * H_C * 4);
  float*  Acum    = (float*) wsalloc(cur, (size_t)B_C * H_C * NC_C * CH_C * 4);
  float*  Alast   = (float*) wsalloc(cur, (size_t)B_C * H_C * NC_C * 4);
  bf16_t* Bb16    = (bf16_t*)wsalloc(cur, (size_t)B_C * NC_C * CH_C * N_C * 2);
  bf16_t* Cb16    = (bf16_t*)wsalloc(cur, (size_t)B_C * NC_C * CH_C * N_C * 2);
  bf16_t* Bt16    = (bf16_t*)wsalloc(cur, (size_t)B_C * NC_C * N_C * CH_C * 2);
  float*  CBt     = (float*) wsalloc(cur, (size_t)B_C * NC_C * CH_C * CH_C * 4);
  float*  states  = (float*) wsalloc(cur, (size_t)B_C * NC_C * H_C * P_C * N_C * 4);
  bf16_t* prev16  = (bf16_t*)wsalloc(cur, (size_t)B_C * NC_C * H_C * P_C * N_C * 2);
  float*  y_g     = (float*) wsalloc(cur, (size_t)B_C * S_C * INTER_C * 4);
  bf16_t* yb      = (bf16_t*)wsalloc(cur, (size_t)B_C * S_C * INTER_C * 2);

  // 1. conversions
  {
    size_t n = (size_t)B_C * S_C * E_C;
    ssm_cvt_x_mask<<<dim3((unsigned)((n + 255) / 256)), dim3(256), 0, stream>>>(x, mask, xb);
  }
  {
    size_t n = (size_t)PROJ_C * E_C;
    ssm_cvt_bf16<<<dim3((unsigned)((n + 255) / 256)), dim3(256), 0, stream>>>(in_w, w_in_b, n);
  }
  {
    size_t n = (size_t)E_C * INTER_C;
    ssm_cvt_bf16<<<dim3((unsigned)((n + 255) / 256)), dim3(256), 0, stream>>>(out_w, w_out_b, n);
  }
  // 2. in_proj GEMM: proj[4096, 8512] = xb @ w_in^T  (ldc padded; 256x128 blocks)
  ssm_gemm_bf16_tn<<<dim3(16, 67, 1), dim3(256), 0, stream>>>(
      xb, w_in_b, proj, B_C * S_C, PROJ_C, E_C, PROJ_LD, 0, 0, 0);
  // 3. conv + silu + mask
  {
    size_t n = (size_t)B_C * S_C * CONVD_C;
    ssm_conv_silu<<<dim3((unsigned)((n + 255) / 256)), dim3(256), 0, stream>>>(
        proj, conv_w, conv_b, mask, hBC);
  }
  // 4. dt + per-chunk cumsum
  ssm_dt_scan<<<dim3(B_C * H_C * NC_C), dim3(CH_C), 0, stream>>>(
      proj, dt_bias, A_log, dt_g, Acum, Alast);
  // 5. bf16 B/C chunk copies + B^T
  ssm_prep_bc<<<dim3(B_C * NC_C), dim3(256), 0, stream>>>(hBC, Bb16, Cb16, Bt16);
  // 6. CB^T batched GEMM (per (b,c)): [256,256] = C[256,128] @ B[256,128]^T
  ssm_gemm_bf16_tn<<<dim3(1, 2, B_C * NC_C), dim3(256), 0, stream>>>(
      Cb16, Bb16, CBt, CH_C, CH_C, N_C, CH_C,
      (long long)CH_C * N_C, (long long)CH_C * N_C, (long long)CH_C * CH_C);
  // 7. per-chunk states
  ssm_chunk_states<<<dim3(B_C * NC_C * H_C), dim3(256), 0, stream>>>(
      hBC, dt_g, Acum, Alast, Bt16, states);
  // 8. inter-chunk recurrence
  {
    size_t n = (size_t)B_C * H_C * P_C * N_C;
    ssm_recurrence<<<dim3((unsigned)(n / 256)), dim3(256), 0, stream>>>(states, Alast, prev16);
  }
  // 9. Y = diag + off + D residual
  ssm_y_kernel<<<dim3(B_C * NC_C * H_C), dim3(256), 0, stream>>>(
      hBC, dt_g, Acum, CBt, prev16, Dv, y_g);
  // 10. gate + rmsnorm
  ssm_gate_norm<<<dim3(B_C * S_C), dim3(256), 0, stream>>>(y_g, proj, norm_w, yb);
  // 11. out_proj GEMM: out[4096, 2048] = yb @ w_out^T
  ssm_gemm_bf16_tn<<<dim3(16, 16, 1), dim3(256), 0, stream>>>(
      yb, w_out_b, out, B_C * S_C, E_C, INTER_C, E_C, 0, 0, 0);
}
